// KMeansModule_43928925504099
// MI455X (gfx1250) — matrix operand order repaired and checked by reading, not compiled, not run on hardware
//
#include <hip/hip_runtime.h>
#include <hip/hip_bf16.h>

#define NPTS   65536     // 16 * 4096
#define DIM    384
#define NCLUST 64
#define KITERS 10
#define KSTEP  32
#define NTILES 4         // 64 clusters / 16 cols per WMMA tile

typedef __attribute__((ext_vector_type(16))) __bf16 v16bf;
typedef __attribute__((ext_vector_type(8)))  float  v8f;

// ---------------------------------------------------------------------------
// Per-cluster prep: convert centers to bf16, compute ||c||^2, zero sums/counts
// grid = 64 blocks (one per cluster), block = 128 threads
// ---------------------------------------------------------------------------
__global__ void prep_centers_kernel(const float* __restrict__ centers,
                                    __bf16* __restrict__ cbf,
                                    float* __restrict__ cnorm,
                                    float* __restrict__ sums,
                                    float* __restrict__ counts) {
    __shared__ float red[128];
    const int c = blockIdx.x;
    const int t = threadIdx.x;
    float acc = 0.f;
    for (int d = t; d < DIM; d += 128) {
        float v = centers[c * DIM + d];
        cbf[c * DIM + d] = (__bf16)v;
        sums[c * DIM + d] = 0.f;
        acc = fmaf(v, v, acc);
    }
    red[t] = acc;
    __syncthreads();
    for (int s = 64; s > 0; s >>= 1) {
        if (t < s) red[t] += red[t + s];
        __syncthreads();
    }
    if (t == 0) { cnorm[c] = red[0]; counts[c] = 0.f; }
}

// ---------------------------------------------------------------------------
// Build one A fragment (16x32 bf16) for K-block kb: lane half g holds
// K {kb+16g..+15}, converted fp32 -> bf16 on the fly.
// ---------------------------------------------------------------------------
__device__ __forceinline__ v16bf load_a_frag(const float* __restrict__ aptr,
                                             int kb, int g) {
    const float4 a0 = *(const float4*)(aptr + kb + 16 * g);
    const float4 a1 = *(const float4*)(aptr + kb + 16 * g + 4);
    const float4 a2 = *(const float4*)(aptr + kb + 16 * g + 8);
    const float4 a3 = *(const float4*)(aptr + kb + 16 * g + 12);
    v16bf a;
    a[0]=(__bf16)a0.x;  a[1]=(__bf16)a0.y;  a[2]=(__bf16)a0.z;  a[3]=(__bf16)a0.w;
    a[4]=(__bf16)a1.x;  a[5]=(__bf16)a1.y;  a[6]=(__bf16)a1.z;  a[7]=(__bf16)a1.w;
    a[8]=(__bf16)a2.x;  a[9]=(__bf16)a2.y;  a[10]=(__bf16)a2.z; a[11]=(__bf16)a2.w;
    a[12]=(__bf16)a3.x; a[13]=(__bf16)a3.y; a[14]=(__bf16)a3.z; a[15]=(__bf16)a3.w;
    return a;
}

// ---------------------------------------------------------------------------
// Assignment (+ fused segment-sum): one wave per 16-point tile.
// Distances d[m,n] = ||c_n||^2 - 2 * <p_m, c_n> via v_wmma_f32_16x16x32_bf16.
// grid = 512 blocks of 256 threads (8 waves) -> 4096 tiles -> 65536 points.
// dynamic LDS: 64*384 sums + 64 counts + 128 labels.
// ---------------------------------------------------------------------------
__global__ void assign_kernel(const float* __restrict__ pts,
                              const __bf16* __restrict__ cbf,
                              const float* __restrict__ cnorm,
                              int* __restrict__ labels,
                              float* __restrict__ sums,
                              float* __restrict__ counts,
                              int doAccum) {
    extern __shared__ float smem[];
    float* s_sums = smem;                         // 64*384 floats
    float* s_cnt  = smem + NCLUST * DIM;          // 64 floats
    int*   s_lab  = (int*)(smem + NCLUST * DIM + NCLUST); // 128 ints

    const int tid  = threadIdx.x;
    const int wave = tid >> 5;
    const int lane = tid & 31;
    const int col  = lane & 15;    // N (and M) index within 16-wide tile
    const int g    = lane >> 4;    // half-wave select

    if (doAccum) {
        for (int i = tid; i < NCLUST * DIM; i += 256) s_sums[i] = 0.f;
        if (tid < NCLUST) s_cnt[tid] = 0.f;
        __syncthreads();
    }

    const int tile    = blockIdx.x * 8 + wave;
    const int rowBase = tile * 16;
    const float* aptr = pts + (size_t)(rowBase + col) * DIM;

    // warm caches for this wave's A rows (global_prefetch_b8)
    __builtin_prefetch(aptr, 0, 1);
    __builtin_prefetch(aptr + DIM / 2, 0, 1);

    v8f acc[NTILES] = {};

    // K-loop, unrolled x2: per step, issue ALL B loads as one clause, then
    // run the 4 WMMAs back-to-back so loads of the next step overlap compute.
    #pragma unroll 2
    for (int kb = 0; kb < DIM; kb += KSTEP) {
        const v16bf a = load_a_frag(aptr, kb, g);

        v16bf bf[NTILES];
        #pragma unroll
        for (int nt = 0; nt < NTILES; ++nt) {
            const int N = nt * 16 + col;
            // B fragment (32x16 bf16): lane holds 16 contiguous K for column N
            bf[nt] = *(const v16bf*)(cbf + (size_t)N * DIM + kb + 16 * g);
        }
        #pragma unroll
        for (int nt = 0; nt < NTILES; ++nt) {
            acc[nt] = __builtin_amdgcn_wmma_f32_16x16x32_bf16(
                false, a, false, bf[nt], (short)0, acc[nt], false, false);
        }
    }

    float cn[NTILES];
    #pragma unroll
    for (int nt = 0; nt < NTILES; ++nt) cn[nt] = cnorm[nt * 16 + col];

    // C/D layout: VGPR r, lane l -> row M = r + 8*(l>>4), col N = l&15
    #pragma unroll
    for (int r = 0; r < 8; ++r) {
        float bestV = 3.4e38f;
        int   bestI = 0;
        #pragma unroll
        for (int nt = 0; nt < NTILES; ++nt) {
            float v = fmaf(-2.f, acc[nt][r], cn[nt]);
            int idx = nt * 16 + col;
            if (v < bestV || (v == bestV && idx < bestI)) { bestV = v; bestI = idx; }
        }
        // min-reduce across 16 lanes of each half-wave (first-min tie-break)
        #pragma unroll
        for (int msk = 1; msk < 16; msk <<= 1) {
            float ov = __shfl_xor(bestV, msk, 32);
            int   oi = __shfl_xor(bestI, msk, 32);
            if (ov < bestV || (ov == bestV && oi < bestI)) { bestV = ov; bestI = oi; }
        }
        if (col == 0) {
            int row = r + 8 * g;
            labels[rowBase + row] = bestI;
            s_lab[wave * 16 + row] = bestI;
        }
    }

    if (doAccum) {
        __syncthreads();
        // LDS-staged segment sums (ds_add_f32), fp32 source points
        for (int r16 = 0; r16 < 16; ++r16) {
            const int lab = s_lab[wave * 16 + r16];
            const float* pp = pts + (size_t)(rowBase + r16) * DIM;
            for (int d = lane; d < DIM; d += 32)
                atomicAdd(&s_sums[lab * DIM + d], pp[d]);
            if (lane == 0) atomicAdd(&s_cnt[lab], 1.0f);
        }
        __syncthreads();
        for (int i = tid; i < NCLUST * DIM; i += 256)
            atomicAdd(&sums[i], s_sums[i]);
        if (tid < NCLUST) atomicAdd(&counts[tid], s_cnt[tid]);
    }
}

// ---------------------------------------------------------------------------
// Centroid update: centers = counts>0 ? sums/max(counts,1) : centers
// ---------------------------------------------------------------------------
__global__ void update_kernel(float* __restrict__ centers,
                              const float* __restrict__ sums,
                              const float* __restrict__ counts) {
    int i = blockIdx.x * blockDim.x + threadIdx.x;
    if (i < NCLUST * DIM) {
        int c = i / DIM;
        float cnt = counts[c];
        if (cnt > 0.f) centers[i] = sums[i] / fmaxf(cnt, 1.f);
    }
}

// ---------------------------------------------------------------------------
// Project the 64 centers through W (fp32 exact): P[c,n] = b[n] + <centers[c,:], W[n,:]>
// grid = (64, 3), block = 128
// ---------------------------------------------------------------------------
__global__ void project_kernel(const float* __restrict__ centers,
                               const float* __restrict__ W,
                               const float* __restrict__ bias,
                               float* __restrict__ P) {
    __shared__ float sc[DIM];
    const int c = blockIdx.x;
    for (int d = threadIdx.x; d < DIM; d += blockDim.x)
        sc[d] = centers[c * DIM + d];
    __syncthreads();
    const int n = blockIdx.y * blockDim.x + threadIdx.x;
    if (n < DIM) {
        float accv = bias[n];
        const float4* wr4 = (const float4*)(W + (size_t)n * DIM);
        #pragma unroll 4
        for (int k4 = 0; k4 < DIM / 4; ++k4) {
            float4 w = wr4[k4];
            int k = k4 * 4;
            accv = fmaf(sc[k + 0], w.x, accv);
            accv = fmaf(sc[k + 1], w.y, accv);
            accv = fmaf(sc[k + 2], w.z, accv);
            accv = fmaf(sc[k + 3], w.w, accv);
        }
        P[c * DIM + n] = accv;
    }
}

// ---------------------------------------------------------------------------
// Gather: out[p,:] = P[labels[p],:]  (float4 vectorized, fully coalesced)
// ---------------------------------------------------------------------------
__global__ void gather_kernel(const float4* __restrict__ P4,
                              const int* __restrict__ labels,
                              float4* __restrict__ out4) {
    const int i = blockIdx.x * blockDim.x + threadIdx.x; // over NPTS * 96
    const int p = i / (DIM / 4);
    const int c = i - p * (DIM / 4);
    out4[i] = P4[(size_t)labels[p] * (DIM / 4) + c];
}

// ---------------------------------------------------------------------------
extern "C" void kernel_launch(void* const* d_in, const int* in_sizes, int n_in,
                              void* d_out, int out_size, void* d_ws, size_t ws_size,
                              hipStream_t stream) {
    const float* x    = (const float*)d_in[0]; // (16,4096,384) fp32
    const float* W    = (const float*)d_in[1]; // (384,384) fp32
    const float* bias = (const float*)d_in[2]; // (384,) fp32
    float* out = (float*)d_out;

    char* ws = (char*)d_ws;
    size_t off = 0;
    float*  centers = (float*)(ws + off);  off += (size_t)NCLUST * DIM * 4;          // 98304
    __bf16* cbf     = (__bf16*)(ws + off); off += (size_t)NCLUST * DIM * 2;          // 49152
    float*  cnorm   = (float*)(ws + off);  off += 256;
    float*  sums    = (float*)(ws + off);  off += (size_t)NCLUST * DIM * 4;          // 98304
    float*  counts  = (float*)(ws + off);  off += 256;
    float*  P       = (float*)(ws + off);  off += (size_t)NCLUST * DIM * 4;          // 98304
    int*    labels  = (int*)(ws + off);    off += (size_t)NPTS * 4;                  // 262144

    // centers0 = points[:64] (contiguous first 64*384 floats of x)
    hipMemcpyAsync(centers, x, (size_t)NCLUST * DIM * sizeof(float),
                   hipMemcpyDeviceToDevice, stream);

    const size_t smem = (size_t)(NCLUST * DIM + NCLUST + 128) * sizeof(float); // ~99 KB
    const int assignBlocks = NPTS / (16 * 8); // 512

    for (int it = 0; it < KITERS; ++it) {
        prep_centers_kernel<<<NCLUST, 128, 0, stream>>>(centers, cbf, cnorm, sums, counts);
        assign_kernel<<<assignBlocks, 256, smem, stream>>>(x, cbf, cnorm, labels,
                                                           sums, counts, 1);
        update_kernel<<<(NCLUST * DIM + 255) / 256, 256, 0, stream>>>(centers, sums, counts);
    }
    // final assignment (labels only)
    prep_centers_kernel<<<NCLUST, 128, 0, stream>>>(centers, cbf, cnorm, sums, counts);
    assign_kernel<<<assignBlocks, 256, smem, stream>>>(x, cbf, cnorm, labels,
                                                       sums, counts, 0);
    // project 64 centers through W (fp32 exact), then gather into output
    project_kernel<<<dim3(NCLUST, 3), 128, 0, stream>>>(centers, W, bias, P);
    gather_kernel<<<(NPTS * (DIM / 4)) / 256, 256, 0, stream>>>(
        (const float4*)P, labels, (float4*)out);
}
